// MeshProcessor_76390288327367
// MI455X (gfx1250) — compile-verified, matrix operand order checked by reading, not run on hardware
//
#include <hip/hip_runtime.h>

#define H_DIM    128
#define N_EDGE   160000
#define N_NODE   10000
#define N_LAYERS 16
#define TM       32      // rows (edges/nodes) per workgroup tile

typedef __attribute__((ext_vector_type(16))) __bf16 v16bf;
typedef __attribute__((ext_vector_type(8)))  float  v8f;

union BfFrag { v16bf v; uint4 q[2]; };

__device__ __forceinline__ unsigned short f2bf(float f) {
    unsigned int u = __float_as_uint(f);
    u += 0x7FFFu + ((u >> 16) & 1u);          // round-to-nearest-even
    return (unsigned short)(u >> 16);
}

// --------------------------------------------------------------------------
// Convert fp32 weights [L][K][128] into bf16 WMMA-B fragment layout:
// per (kb,nb) 32x16 tile: 32 lanes x 16 bf16 contiguous per lane, element e of
// lane l maps to k = kb*32 + (e>=8?16:0) + (l>=16?8:0) + ((e>>1)&3)*2 + (e&1),
// n = nb*16 + (l&15)  (ISA 7.12.2 16-bit operand layout).
// --------------------------------------------------------------------------
__global__ void swizzle_weights_kernel(const float* __restrict__ W,
                                       unsigned short* __restrict__ out,
                                       int K, int total) {
    int idx = blockIdx.x * blockDim.x + threadIdx.x;
    if (idx >= total) return;
    int per   = K * H_DIM;
    int layer = idx / per;
    int r     = idx - layer * per;
    int tile  = r >> 9;          // /512
    int rem   = r & 511;
    int l     = rem >> 4;
    int e     = rem & 15;
    int kb    = tile >> 3;
    int nb    = tile & 7;
    int k = kb * 32 + ((e >= 8) ? 16 : 0) + ((l >= 16) ? 8 : 0)
          + (((e >> 1) & 3) << 1) + (e & 1);
    int n = nb * 16 + (l & 15);
    out[idx] = f2bf(W[(size_t)layer * per + (size_t)k * H_DIM + n]);
}

__global__ void zero_kernel(float* __restrict__ p, int n) {
    int i = blockIdx.x * blockDim.x + threadIdx.x;
    if (i < n) p[i] = 0.f;
}

__global__ void init_feats_kernel(const float* __restrict__ ef_in,
                                  const float* __restrict__ nf_in,
                                  float* __restrict__ out) {
    int i = blockIdx.x * blockDim.x + threadIdx.x;   // float4 index
    const int e4 = N_EDGE * H_DIM / 4;
    const int n4 = N_NODE * H_DIM / 4;
    float4* o = (float4*)out;
    if (i < e4)            o[i] = ((const float4*)ef_in)[i];
    else if (i < e4 + n4)  o[i] = ((const float4*)nf_in)[i - e4];
}

// --------------------------------------------------------------------------
// Fused MLP layer: gather/concat -> bf16 -> WMMA GEMM1 -> SiLU -> WMMA GEMM2
// -> LayerNorm -> residual; edge mode additionally scatter-adds into agg.
// Block: 256 threads = 8 wave32; tile: TM=32 rows x 128 cols.
// Wave w owns output tiles (mi = w>>2, nb = (w&3)*2 .. +1).
// --------------------------------------------------------------------------
template<int K1, bool IS_EDGE>
__global__ __launch_bounds__(256) void mlp_layer_kernel(
    float* __restrict__ feat,                  // ef or nf: residual in/out
    const float* __restrict__ nf,              // node features (gather source)
    const float* __restrict__ agg_in,          // node mode: aggregated edge sums
    const int* __restrict__ src,
    const int* __restrict__ dst,
    const unsigned short* __restrict__ w1s,    // bf16 swizzled [K1 x 128]
    const float* __restrict__ b1,
    const unsigned short* __restrict__ w2s,    // bf16 swizzled [128 x 128]
    const float* __restrict__ b2,
    const float* __restrict__ gamma,
    const float* __restrict__ beta,
    float* __restrict__ agg_out,               // edge mode: atomic target
    int nRows) {

    __shared__ __align__(16) unsigned short shA[TM * K1];      // concat input, bf16
    __shared__ __align__(16) unsigned short shH[TM * H_DIM];   // hidden, bf16
    __shared__ __align__(16) float          shY[TM * H_DIM];   // pre-LN output, f32
    __shared__ float shPart[TM * 8 * 2];
    __shared__ float shMu[TM], shRs[TM];
    __shared__ int   shDst[TM], shSrc[TM];

    const int tid  = threadIdx.x;
    const int lane = tid & 31;
    const int wave = tid >> 5;
    const int m0   = blockIdx.x * TM;

    if (IS_EDGE) {
        if (tid < TM) {
            int e = m0 + tid;
            shSrc[tid] = (e < nRows) ? src[e] : 0;
            shDst[tid] = (e < nRows) ? dst[e] : 0;
        }
        __syncthreads();
    }

    // ---- Phase 1: gather + fp32->bf16 into LDS (float4 granularity) ----
    for (int idx = tid; idx < TM * (K1 / 4); idx += 256) {
        int r   = idx / (K1 / 4);
        int c   = (idx - r * (K1 / 4)) * 4;
        int row = m0 + r;
        float4 v = make_float4(0.f, 0.f, 0.f, 0.f);
        if (row < nRows) {
            const float* p;
            if (IS_EDGE) {
                if (c < H_DIM)          p = feat   + (size_t)row      * H_DIM + c;
                else if (c < 2 * H_DIM) p = nf     + (size_t)shSrc[r] * H_DIM + (c - H_DIM);
                else                    p = nf     + (size_t)shDst[r] * H_DIM + (c - 2 * H_DIM);
            } else {
                if (c < H_DIM)          p = agg_in + (size_t)row * H_DIM + c;
                else                    p = nf     + (size_t)row * H_DIM + (c - H_DIM);
            }
            v = *(const float4*)p;
        }
        ushort4 b;
        b.x = f2bf(v.x); b.y = f2bf(v.y); b.z = f2bf(v.z); b.w = f2bf(v.w);
        *(ushort4*)&shA[r * K1 + c] = b;
    }
    __syncthreads();

    const int mi   = wave >> 2;                       // 0..1  (16-row block)
    const int nb0  = (wave & 3) * 2;                  // 0,2,4,6 (16-col blocks)
    const int ncol = lane & 15;
    const int mrow = mi * 16 + ((lane >= 16) ? 8 : 0);
    const int arow = mi * 16 + ncol;

    // ---- Phase 2: GEMM1 (K1 -> 128), bf16 WMMA, fp32 accumulate ----
    v8f acc0 = {}; v8f acc1 = {};
    #pragma unroll
    for (int kb = 0; kb < K1 / 32; ++kb) {
        BfFrag a, bfr0, bfr1;
        int kbase = kb * 32 + ((lane >= 16) ? 8 : 0);
        a.q[0] = *(const uint4*)&shA[arow * K1 + kbase];
        a.q[1] = *(const uint4*)&shA[arow * K1 + kbase + 16];
        const uint4* bp0 = (const uint4*)&w1s[((kb * 8 + nb0    ) * 32 + lane) * 16];
        const uint4* bp1 = (const uint4*)&w1s[((kb * 8 + nb0 + 1) * 32 + lane) * 16];
        bfr0.q[0] = bp0[0]; bfr0.q[1] = bp0[1];
        bfr1.q[0] = bp1[0]; bfr1.q[1] = bp1[1];
        acc0 = __builtin_amdgcn_wmma_f32_16x16x32_bf16(false, a.v, false, bfr0.v,
                                                       (short)0, acc0, false, false);
        acc1 = __builtin_amdgcn_wmma_f32_16x16x32_bf16(false, a.v, false, bfr1.v,
                                                       (short)0, acc1, false, false);
    }

    // bias + SiLU, re-quantize hidden to bf16 in LDS
    {
        int n0 = nb0 * 16 + ncol;
        float bias0 = b1[n0], bias1 = b1[n0 + 16];
        #pragma unroll
        for (int v = 0; v < 8; ++v) {
            float x0 = acc0[v] + bias0;
            float x1 = acc1[v] + bias1;
            x0 = x0 / (1.f + __expf(-x0));
            x1 = x1 / (1.f + __expf(-x1));
            shH[(mrow + v) * H_DIM + n0]      = f2bf(x0);
            shH[(mrow + v) * H_DIM + n0 + 16] = f2bf(x1);
        }
    }
    __syncthreads();

    // ---- Phase 3: GEMM2 (128 -> 128) ----
    v8f y0 = {}; v8f y1 = {};
    #pragma unroll
    for (int kb = 0; kb < H_DIM / 32; ++kb) {
        BfFrag a, bfr0, bfr1;
        int kbase = kb * 32 + ((lane >= 16) ? 8 : 0);
        a.q[0] = *(const uint4*)&shH[arow * H_DIM + kbase];
        a.q[1] = *(const uint4*)&shH[arow * H_DIM + kbase + 16];
        const uint4* bp0 = (const uint4*)&w2s[((kb * 8 + nb0    ) * 32 + lane) * 16];
        const uint4* bp1 = (const uint4*)&w2s[((kb * 8 + nb0 + 1) * 32 + lane) * 16];
        bfr0.q[0] = bp0[0]; bfr0.q[1] = bp0[1];
        bfr1.q[0] = bp1[0]; bfr1.q[1] = bp1[1];
        y0 = __builtin_amdgcn_wmma_f32_16x16x32_bf16(false, a.v, false, bfr0.v,
                                                     (short)0, y0, false, false);
        y1 = __builtin_amdgcn_wmma_f32_16x16x32_bf16(false, a.v, false, bfr1.v,
                                                     (short)0, y1, false, false);
    }
    {
        int n0 = nb0 * 16 + ncol;
        float bias0 = b2[n0], bias1 = b2[n0 + 16];
        #pragma unroll
        for (int v = 0; v < 8; ++v) {
            shY[(mrow + v) * H_DIM + n0]      = y0[v] + bias0;
            shY[(mrow + v) * H_DIM + n0 + 16] = y1[v] + bias1;
        }
    }
    __syncthreads();

    // ---- Phase 4: LayerNorm (row-wise over 128) ----
    {
        int r = tid >> 3, t = tid & 7;                 // 8 threads per row
        const float* yr = &shY[r * H_DIM + t * 16];
        float s = 0.f, ss = 0.f;
        #pragma unroll
        for (int i = 0; i < 16; ++i) { float x = yr[i]; s += x; ss += x * x; }
        shPart[r * 8 + t]          = s;
        shPart[TM * 8 + r * 8 + t] = ss;
    }
    __syncthreads();
    if (tid < TM) {
        float s = 0.f, ss = 0.f;
        #pragma unroll
        for (int t = 0; t < 8; ++t) {
            s  += shPart[tid * 8 + t];
            ss += shPart[TM * 8 + tid * 8 + t];
        }
        float mu  = s * (1.f / H_DIM);
        float var = ss * (1.f / H_DIM) - mu * mu;
        shMu[tid] = mu;
        shRs[tid] = rsqrtf(var + 1e-5f);
    }
    __syncthreads();

    // ---- Phase 5: affine + residual (+ fused scatter-sum for edges) ----
    for (int idx = tid; idx < TM * H_DIM; idx += 256) {
        int r   = idx >> 7;
        int n   = idx & (H_DIM - 1);
        int row = m0 + r;
        if (row < nRows) {
            float val = (shY[idx] - shMu[r]) * shRs[r] * gamma[n] + beta[n];
            size_t off = (size_t)row * H_DIM + n;
            float o = feat[off] + val;
            feat[off] = o;
            if (IS_EDGE) atomicAdd(&agg_out[(size_t)shDst[r] * H_DIM + n], o);
        }
    }
}

// --------------------------------------------------------------------------
extern "C" void kernel_launch(void* const* d_in, const int* in_sizes, int n_in,
                              void* d_out, int out_size, void* d_ws, size_t ws_size,
                              hipStream_t stream) {
    (void)in_sizes; (void)n_in; (void)out_size; (void)ws_size;

    const float* efeat = (const float*)d_in[0];
    const float* nfeat = (const float*)d_in[1];
    const int*   src   = (const int*)d_in[2];
    const int*   dst   = (const int*)d_in[3];
    const float* eW1   = (const float*)d_in[4];
    const float* eb1   = (const float*)d_in[5];
    const float* eW2   = (const float*)d_in[6];
    const float* eb2   = (const float*)d_in[7];
    const float* eg    = (const float*)d_in[8];
    const float* ebt   = (const float*)d_in[9];
    const float* nW1   = (const float*)d_in[10];
    const float* nb1   = (const float*)d_in[11];
    const float* nW2   = (const float*)d_in[12];
    const float* nb2   = (const float*)d_in[13];
    const float* ng    = (const float*)d_in[14];
    const float* nbt   = (const float*)d_in[15];

    // ef / nf live in d_out (updated in place across layers -> final = output)
    float* ef = (float*)d_out;
    float* nf = ef + (size_t)N_EDGE * H_DIM;

    // workspace: agg + bf16-swizzled weights
    unsigned char* ws = (unsigned char*)d_ws;
    float* agg = (float*)ws;                 ws += (size_t)N_NODE * H_DIM * sizeof(float);
    unsigned short* eW1s = (unsigned short*)ws; ws += (size_t)N_LAYERS * 3 * H_DIM * H_DIM * 2;
    unsigned short* eW2s = (unsigned short*)ws; ws += (size_t)N_LAYERS *     H_DIM * H_DIM * 2;
    unsigned short* nW1s = (unsigned short*)ws; ws += (size_t)N_LAYERS * 2 * H_DIM * H_DIM * 2;
    unsigned short* nW2s = (unsigned short*)ws; ws += (size_t)N_LAYERS *     H_DIM * H_DIM * 2;

    // one-time (per launch) weight conversion + feature init
    {
        int t = N_LAYERS * 3 * H_DIM * H_DIM;
        swizzle_weights_kernel<<<(t + 255) / 256, 256, 0, stream>>>(eW1, eW1s, 3 * H_DIM, t);
        t = N_LAYERS * H_DIM * H_DIM;
        swizzle_weights_kernel<<<(t + 255) / 256, 256, 0, stream>>>(eW2, eW2s, H_DIM, t);
        t = N_LAYERS * 2 * H_DIM * H_DIM;
        swizzle_weights_kernel<<<(t + 255) / 256, 256, 0, stream>>>(nW1, nW1s, 2 * H_DIM, t);
        t = N_LAYERS * H_DIM * H_DIM;
        swizzle_weights_kernel<<<(t + 255) / 256, 256, 0, stream>>>(nW2, nW2s, H_DIM, t);

        int t4 = (N_EDGE + N_NODE) * H_DIM / 4;
        init_feats_kernel<<<(t4 + 255) / 256, 256, 0, stream>>>(efeat, nfeat, (float*)d_out);
    }

    const int edgeBlocks = (N_EDGE + TM - 1) / TM;   // 5000
    const int nodeBlocks = (N_NODE + TM - 1) / TM;   // 313
    const int aggN       = N_NODE * H_DIM;

    for (int l = 0; l < N_LAYERS; ++l) {
        zero_kernel<<<(aggN + 255) / 256, 256, 0, stream>>>(agg, aggN);

        mlp_layer_kernel<3 * H_DIM, true><<<edgeBlocks, 256, 0, stream>>>(
            ef, nf, nullptr, src, dst,
            eW1s + (size_t)l * 3 * H_DIM * H_DIM, eb1 + l * H_DIM,
            eW2s + (size_t)l *     H_DIM * H_DIM, eb2 + l * H_DIM,
            eg + l * H_DIM, ebt + l * H_DIM,
            agg, N_EDGE);

        mlp_layer_kernel<2 * H_DIM, false><<<nodeBlocks, 256, 0, stream>>>(
            nf, nf, agg, nullptr, nullptr,
            nW1s + (size_t)l * 2 * H_DIM * H_DIM, nb1 + l * H_DIM,
            nW2s + (size_t)l *     H_DIM * H_DIM, nb2 + l * H_DIM,
            ng + l * H_DIM, nbt + l * H_DIM,
            nullptr, N_NODE);
    }
}
